// mutliHeadAttention_38525856645418
// MI455X (gfx1250) — matrix operand order repaired and checked by reading, not compiled, not run on hardware
//
#include <hip/hip_runtime.h>

// ---------------------------------------------------------------------------
// MI455X (gfx1250) fused multi-head attention:
//   Q/K/V proj (bf16 WMMA) -> flash attention (bf16 WMMA, online softmax,
//   K panel staged by the Tensor Data Mover) -> O proj + residual (bf16 WMMA,
//   f32 out) -> LayerNorm (in-place).
// Shapes: B=4, M=2048, D=1024, H=16, Dh=64.  ~137 GFLOP, compute-bound on the
// WMMA pipes -> all matmuls run on v_wmma_f32_16x16x32_bf16 (f32 accum).
// ---------------------------------------------------------------------------

typedef unsigned short u16;
typedef unsigned int   u32;
typedef unsigned long long u64;
typedef __attribute__((ext_vector_type(16))) __bf16 v16bf;
typedef __attribute__((ext_vector_type(8)))  float  v8f;
typedef __attribute__((ext_vector_type(4)))  u32    v4u;
typedef __attribute__((ext_vector_type(8)))  u32    v8u;

#define BDIM   4
#define MDIM   2048
#define DMODEL 1024
#define NHEAD  16
#define DHEAD  64
#define BM     (BDIM * MDIM)      // 8192 rows
#define NEGV   (-1000000000.0f)

union Frag {
  v16bf v;
  uint4 q[2];
  u16   s[16];
};

static __device__ inline u16 f2bf(float f) {
  u32 u = __float_as_uint(f);
  u32 r = u + 0x7FFFu + ((u >> 16) & 1u);   // round-to-nearest-even
  return (u16)(r >> 16);
}

static __device__ inline v8f wmma_bf16f32(v16bf a, v16bf b, v8f c) {
  // (neg_a, A, neg_b, B, c_mod, C, reuse_a, reuse_b)
  return __builtin_amdgcn_wmma_f32_16x16x32_bf16(false, a, false, b,
                                                 (short)0, c, false, false);
}

// Raw LDS byte offset of a __shared__ object (what D#.lds_addr wants).
static __device__ inline u32 lds_off(void* p) {
  return (u32)(__UINTPTR_TYPE__)(__attribute__((address_space(3))) void*)p;
}

// xor-lane exchange inside the 32-lane wave (masks <= 8 stay within the
// 16-lane halves that hold one C-tile row).
#define XORF(x, m) \
  __int_as_float(__builtin_amdgcn_ds_swizzle(__float_as_int(x), (((m) << 10) | 0x1f)))

// ---------------------------------------------------------------------------
// Stage 0: f32 -> bf16 conversion (packs 2 elements per 32-bit store)
// ---------------------------------------------------------------------------
__global__ __launch_bounds__(256) void cvt_bf16_kernel(const float* __restrict__ src,
                                                       u16* __restrict__ dst,
                                                       int npairs) {
  int i = blockIdx.x * blockDim.x + threadIdx.x;
  if (i < npairs) {
    float a = src[2 * i + 0];
    float b = src[2 * i + 1];
    u32 p = (u32)f2bf(a) | ((u32)f2bf(b) << 16);
    ((u32*)dst)[i] = p;
  }
}

// ---------------------------------------------------------------------------
// Stage 1: projection GEMM  Y[m,n] = sum_k X[m,k] * W[n,k]   (y = x @ W.T)
// WG tile 128x128, 8 waves, wave tile 32x64 (2 M-tiles x 4 N-tiles).
// Output scattered head-major: Yh[b][h][m][d], bf16, optional scale (Q: 1/8).
// ---------------------------------------------------------------------------
__global__ __launch_bounds__(256) void proj_qkv_kernel(const u16* __restrict__ X,
                                                       const u16* __restrict__ W,
                                                       u16* __restrict__ Yh,
                                                       float scale) {
  const int lane = threadIdx.x & 31;
  const int wave = threadIdx.x >> 5;
  const int r = lane & 15, hi = lane >> 4;
  const int mWave = blockIdx.x * 128 + (wave >> 1) * 32;
  const int nWave = blockIdx.y * 128 + (wave & 1) * 64;

  v8f acc[2][4] = {};

  for (int k0 = 0; k0 < DMODEL; k0 += 32) {
    Frag a[2], bw[4];
#pragma unroll
    for (int mt = 0; mt < 2; ++mt) {
      const uint4* p = (const uint4*)(X + (size_t)(mWave + mt * 16 + r) * DMODEL + k0 + hi * 8);
      a[mt].q[0] = p[0];      // K = c0..c0+7
      a[mt].q[1] = p[2];      // K = c0+16..c0+23
    }
#pragma unroll
    for (int nt = 0; nt < 4; ++nt) {
      const uint4* p = (const uint4*)(W + (size_t)(nWave + nt * 16 + r) * DMODEL + k0 + hi * 8);
      bw[nt].q[0] = p[0];     // K = c..c+7
      bw[nt].q[1] = p[2];     // K = c+16..c+23
    }
    if (k0 + 32 < DMODEL) {   // hint next K-step into cache (global_prefetch_b8)
      __builtin_prefetch(X + (size_t)(mWave + r) * DMODEL + k0 + 32, 0, 3);
      __builtin_prefetch(W + (size_t)(nWave + r) * DMODEL + k0 + 32, 0, 3);
    }
#pragma unroll
    for (int mt = 0; mt < 2; ++mt)
#pragma unroll
      for (int nt = 0; nt < 4; ++nt)
        acc[mt][nt] = wmma_bf16f32(a[mt].v, bw[nt].v, acc[mt][nt]);
  }

  // C layout: lane L holds col = (L&15), rows v + 8*(L>>4) in VGPR v.
#pragma unroll
  for (int mt = 0; mt < 2; ++mt)
#pragma unroll
    for (int nt = 0; nt < 4; ++nt)
#pragma unroll
      for (int v = 0; v < 8; ++v) {
        int row = mWave + mt * 16 + hi * 8 + v;     // global token row in [0, BM)
        int col = nWave + nt * 16 + r;              // feature in [0, D_FF)
        int bb = row >> 11, m = row & (MDIM - 1);
        int h = col >> 6, d = col & (DHEAD - 1);
        Yh[(((size_t)(bb * NHEAD + h)) * MDIM + m) * DHEAD + d] =
            f2bf(acc[mt][nt][v] * scale);
      }
}

// ---------------------------------------------------------------------------
// Stage 2: flash attention.  One WG = 4 waves = 128 q-rows of one (b,h).
// Streams K/V in 32-wide chunks.  The shared K panel is DMA'd into LDS by the
// Tensor Data Mover (tensor_load_to_lds, padded rows: 128B + 16B pad -> 72
// elem pitch, bank-conflict friendly), V is staged transposed in LDS, and P
// round-trips through LDS to convert C-layout -> A-layout for the ctx WMMA.
// ---------------------------------------------------------------------------
__global__ __launch_bounds__(128) void flash_attn_kernel(const u16* __restrict__ Qh,
                                                         const u16* __restrict__ Kh,
                                                         const u16* __restrict__ Vh,
                                                         const unsigned char* __restrict__ maskp,
                                                         u16* __restrict__ Cx) {
  __shared__ u16 Ks[32][72];         // K chunk: [kv][d], TDM-padded pitch 144B
  __shared__ u16 Vt[DHEAD][40];      // transposed V chunk: [d][kv], padded
  __shared__ u16 Pl[4][32][40];      // per-wave P tile: [q][kv], padded

  const int lane = threadIdx.x & 31;
  const int wave = threadIdx.x >> 5;
  const int r = lane & 15, hi = lane >> 4;
  const int bh = blockIdx.y;
  const int bb = bh >> 4;            // batch
  const int h  = bh & 15;            // head
  const int qWave = blockIdx.x * 128 + wave * 32;

  const u16* Qp = Qh + (size_t)bh * MDIM * DHEAD;
  const u16* Kp = Kh + (size_t)bh * MDIM * DHEAD;
  const u16* Vp = Vh + (size_t)bh * MDIM * DHEAD;

  // Preload Q fragments (A layout), reused for all kv chunks.
  Frag qf[2][2];
#pragma unroll
  for (int mt = 0; mt < 2; ++mt)
#pragma unroll
    for (int ds = 0; ds < 2; ++ds) {
      const uint4* p = (const uint4*)(Qp + (size_t)(qWave + mt * 16 + r) * DHEAD + ds * 32 + hi * 8);
      qf[mt][ds].q[0] = p[0];
      qf[mt][ds].q[1] = p[2];
    }

  const u32 ksLds = lds_off(&Ks[0][0]);

  float mst[2][8], lst[2][8];
#pragma unroll
  for (int mt = 0; mt < 2; ++mt)
#pragma unroll
    for (int v = 0; v < 8; ++v) { mst[mt][v] = -3.0e38f; lst[mt][v] = 0.0f; }
  v8f ctx[2][4] = {};

  for (int kv0 = 0; kv0 < MDIM; kv0 += 32) {
    __syncthreads();   // previous chunk's LDS reads complete

    // --- Wave 0: TDM DMA of the K panel (32 kv rows x 64 d) into Ks. ------
    if (wave == 0) {
      u64 ga = (u64)(__UINTPTR_TYPE__)(Kp + (size_t)kv0 * DHEAD);
      // D# group 0: count=1, lds_addr, global_addr, type=2 ("image")
      v4u g0 = { 1u,
                 ksLds,
                 (u32)ga,
                 ((u32)(ga >> 32) & 0x01FFFFFFu) | 0x80000000u };
      // D# group 1: data_size=2B, pad_enable, pad_interval=32DW, pad_amount=4DW,
      // tensor_dim0=64, tensor_dim1=32, tile=64x32, stride0=64.
      v4u z = { 0u, 0u, 0u, 0u };
      v8u g1 = { (1u << 16) | (1u << 20) | (4u << 22) | (3u << 25),
                 (64u << 16),            // tensor_dim0 = 64 (bits 79:48 lo)
                 (32u << 16),            // tensor_dim1 = 32 (bits 111:80 lo)
                 (64u << 16),            // tile_dim0 = 64   (bits 127:112)
                 32u,                    // tile_dim1 = 32   (bits 143:128)
                 64u,                    // tensor_dim0_stride = 64 (bits 207:160 lo)
                 0u, 0u };
      asm volatile("tensor_load_to_lds %0, %1, %2, %3"
                   :: "s"(g0), "s"(g1), "s"(z), "s"(z)
                   : "memory");
      __builtin_amdgcn_s_wait_tensorcnt(0);
    }

    // --- All threads: stage V chunk transposed: Vt[d][kv]. -----------------
    {
      const int kvr = threadIdx.x >> 2;        // 0..31
      const int dp  = (threadIdx.x & 3) * 16;  // 0,16,32,48
      Frag vl;
      const uint4* p = (const uint4*)(Vp + (size_t)(kv0 + kvr) * DHEAD + dp);
      vl.q[0] = p[0]; vl.q[1] = p[1];
#pragma unroll
      for (int j = 0; j < 16; ++j) Vt[dp + j][kvr] = vl.s[j];
    }
    __syncthreads();   // Ks (TDM) + Vt visible to all waves

#pragma unroll
    for (int mt = 0; mt < 2; ++mt) {
      // S = Q . K^T for two 16-wide kv tiles; K-fragments from the TDM panel
      v8f st[2];
#pragma unroll
      for (int kt = 0; kt < 2; ++kt) {
        v8f a = {};
#pragma unroll
        for (int ds = 0; ds < 2; ++ds) {
          Frag kf;
          const uint4* p = (const uint4*)&Ks[kt * 16 + r][ds * 32 + hi * 8];
          kf.q[0] = p[0]; kf.q[1] = p[2];
          a = wmma_bf16f32(qf[mt][ds].v, kf.v, a);
        }
        st[kt] = a;
      }
      // Online softmax (row lives across the 16-lane half; v indexes 8 rows)
      float alpha[8];
      const int qrowBase = qWave + mt * 16 + hi * 8;
#pragma unroll
      for (int v = 0; v < 8; ++v) {
        float s0 = st[0][v], s1 = st[1][v];
        size_t mrow = ((size_t)bb * MDIM + (qrowBase + v)) * MDIM + kv0;
        if (maskp[mrow + r])      s0 = NEGV;
        if (maskp[mrow + 16 + r]) s1 = NEGV;
        float cm = fmaxf(s0, s1);
        cm = fmaxf(cm, XORF(cm, 1));
        cm = fmaxf(cm, XORF(cm, 2));
        cm = fmaxf(cm, XORF(cm, 4));
        cm = fmaxf(cm, XORF(cm, 8));
        float mold = mst[mt][v];
        float mnew = fmaxf(mold, cm);
        float al = __expf(mold - mnew);
        float p0 = __expf(s0 - mnew);
        float p1 = __expf(s1 - mnew);
        float ps = p0 + p1;
        ps += XORF(ps, 1);
        ps += XORF(ps, 2);
        ps += XORF(ps, 4);
        ps += XORF(ps, 8);
        lst[mt][v] = lst[mt][v] * al + ps;
        mst[mt][v] = mnew;
        alpha[v] = al;
        Pl[wave][mt * 16 + hi * 8 + v][r]      = f2bf(p0);
        Pl[wave][mt * 16 + hi * 8 + v][16 + r] = f2bf(p1);
      }
#pragma unroll
      for (int nt = 0; nt < 4; ++nt)
#pragma unroll
        for (int v = 0; v < 8; ++v) ctx[mt][nt][v] *= alpha[v];
    }
    __syncthreads();   // P tiles visible (and Ks/Vt reads ordered)

    // ctx += P(32xK=32) . V(32x64): A from Pl (A layout), B from Vt (B layout)
#pragma unroll
    for (int mt = 0; mt < 2; ++mt) {
      Frag pa;
      const uint4* p = (const uint4*)&Pl[wave][mt * 16 + r][hi * 8];
      pa.q[0] = p[0]; pa.q[1] = p[2];
#pragma unroll
      for (int nt = 0; nt < 4; ++nt) {
        Frag vb;
        const uint4* q = (const uint4*)&Vt[nt * 16 + r][hi * 8];
        vb.q[0] = q[0]; vb.q[1] = q[2];
        ctx[mt][nt] = wmma_bf16f32(pa.v, vb.v, ctx[mt][nt]);
      }
    }
  }

  // Finalize: ctx /= l, store bf16 into [B, M, D_FF] (heads re-interleaved)
#pragma unroll
  for (int mt = 0; mt < 2; ++mt) {
    float rinv[8];
#pragma unroll
    for (int v = 0; v < 8; ++v) rinv[v] = 1.0f / fmaxf(lst[mt][v], 1e-20f);
#pragma unroll
    for (int nt = 0; nt < 4; ++nt)
#pragma unroll
      for (int v = 0; v < 8; ++v) {
        int row = qWave + mt * 16 + hi * 8 + v;
        int col = h * DHEAD + nt * 16 + r;
        Cx[((size_t)bb * MDIM + row) * DMODEL + col] = f2bf(ctx[mt][nt][v] * rinv[v]);
      }
  }
}

// ---------------------------------------------------------------------------
// Stage 3: output projection + residual.  out = Ctx @ Wo.T + input_Q  (f32)
// ---------------------------------------------------------------------------
__global__ __launch_bounds__(256) void oproj_kernel(const u16* __restrict__ Cx,
                                                    const u16* __restrict__ Wo,
                                                    const float* __restrict__ resid,
                                                    float* __restrict__ out) {
  const int lane = threadIdx.x & 31;
  const int wave = threadIdx.x >> 5;
  const int r = lane & 15, hi = lane >> 4;
  const int mWave = blockIdx.x * 128 + (wave >> 1) * 32;
  const int nWave = blockIdx.y * 128 + (wave & 1) * 64;

  v8f acc[2][4] = {};
  for (int k0 = 0; k0 < DMODEL; k0 += 32) {
    Frag a[2], bw[4];
#pragma unroll
    for (int mt = 0; mt < 2; ++mt) {
      const uint4* p = (const uint4*)(Cx + (size_t)(mWave + mt * 16 + r) * DMODEL + k0 + hi * 8);
      a[mt].q[0] = p[0]; a[mt].q[1] = p[2];
    }
#pragma unroll
    for (int nt = 0; nt < 4; ++nt) {
      const uint4* p = (const uint4*)(Wo + (size_t)(nWave + nt * 16 + r) * DMODEL + k0 + hi * 8);
      bw[nt].q[0] = p[0]; bw[nt].q[1] = p[2];
    }
    if (k0 + 32 < DMODEL) {
      __builtin_prefetch(Cx + (size_t)(mWave + r) * DMODEL + k0 + 32, 0, 3);
      __builtin_prefetch(Wo + (size_t)(nWave + r) * DMODEL + k0 + 32, 0, 3);
    }
#pragma unroll
    for (int mt = 0; mt < 2; ++mt)
#pragma unroll
      for (int nt = 0; nt < 4; ++nt)
        acc[mt][nt] = wmma_bf16f32(a[mt].v, bw[nt].v, acc[mt][nt]);
  }

#pragma unroll
  for (int mt = 0; mt < 2; ++mt)
#pragma unroll
    for (int nt = 0; nt < 4; ++nt)
#pragma unroll
      for (int v = 0; v < 8; ++v) {
        int row = mWave + mt * 16 + hi * 8 + v;
        int col = nWave + nt * 16 + r;
        size_t idx = (size_t)row * DMODEL + col;
        out[idx] = acc[mt][nt][v] + resid[idx];
      }
}

// ---------------------------------------------------------------------------
// Stage 4: LayerNorm in place, one block per token row of 1024.
// ---------------------------------------------------------------------------
__global__ __launch_bounds__(256) void layernorm_kernel(float* __restrict__ out,
                                                        const float* __restrict__ g,
                                                        const float* __restrict__ bta) {
  __shared__ float red[256];
  const int row = blockIdx.x;
  const int t = threadIdx.x;
  float x[4];
  float s = 0.f, ss = 0.f;
#pragma unroll
  for (int i = 0; i < 4; ++i) {
    x[i] = out[(size_t)row * DMODEL + t + i * 256];
    s += x[i];
    ss += x[i] * x[i];
  }
  red[t] = s;
  __syncthreads();
  for (int k = 128; k > 0; k >>= 1) { if (t < k) red[t] += red[t + k]; __syncthreads(); }
  float mu = red[0] * (1.0f / DMODEL);
  __syncthreads();
  red[t] = ss;
  __syncthreads();
  for (int k = 128; k > 0; k >>= 1) { if (t < k) red[t] += red[t + k]; __syncthreads(); }
  float var = red[0] * (1.0f / DMODEL) - mu * mu;
  float rstd = rsqrtf(var + 1e-5f);
#pragma unroll
  for (int i = 0; i < 4; ++i) {
    int c = t + i * 256;
    out[(size_t)row * DMODEL + c] = (x[i] - mu) * rstd * g[c] + bta[c];
  }
}

// ---------------------------------------------------------------------------
// Host launcher
// ---------------------------------------------------------------------------
extern "C" void kernel_launch(void* const* d_in, const int* in_sizes, int n_in,
                              void* d_out, int out_size, void* d_ws, size_t ws_size,
                              hipStream_t stream) {
  const float* inQ = (const float*)d_in[0];
  const float* inK = (const float*)d_in[1];
  const float* inV = (const float*)d_in[2];
  const unsigned char* mask = (const unsigned char*)d_in[3];
  const float* wq = (const float*)d_in[4];
  const float* wk = (const float*)d_in[5];
  const float* wv = (const float*)d_in[6];
  const float* wo = (const float*)d_in[7];
  const float* gamma = (const float*)d_in[8];
  const float* beta  = (const float*)d_in[9];
  float* out = (float*)d_out;

  char* ws = (char*)d_ws;
  const size_t SZX = (size_t)BM * DMODEL * sizeof(u16);       // 16 MB
  const size_t SZW = (size_t)DMODEL * DMODEL * sizeof(u16);   // 2 MB
  u16* Xq = (u16*)(ws + 0 * SZX);
  u16* Xk = (u16*)(ws + 1 * SZX);
  u16* Xv = (u16*)(ws + 2 * SZX);
  u16* Wq = (u16*)(ws + 3 * SZX + 0 * SZW);
  u16* Wk = (u16*)(ws + 3 * SZX + 1 * SZW);
  u16* Wv = (u16*)(ws + 3 * SZX + 2 * SZW);
  u16* Wo = (u16*)(ws + 3 * SZX + 3 * SZW);
  u16* Qh = (u16*)(ws + 3 * SZX + 4 * SZW);
  u16* Kh = (u16*)(ws + 4 * SZX + 4 * SZW);
  u16* Vh = (u16*)(ws + 5 * SZX + 4 * SZW);
  u16* Cx = (u16*)(ws + 6 * SZX + 4 * SZW);

  // Stage 0: bf16 conversions
  {
    const int nx = BM * DMODEL;      // 8,388,608
    const int nw = DMODEL * DMODEL;  // 1,048,576
    cvt_bf16_kernel<<<nx / 512, 256, 0, stream>>>(inQ, Xq, nx / 2);
    cvt_bf16_kernel<<<nx / 512, 256, 0, stream>>>(inK, Xk, nx / 2);
    cvt_bf16_kernel<<<nx / 512, 256, 0, stream>>>(inV, Xv, nx / 2);
    cvt_bf16_kernel<<<nw / 512, 256, 0, stream>>>(wq, Wq, nw / 2);
    cvt_bf16_kernel<<<nw / 512, 256, 0, stream>>>(wk, Wk, nw / 2);
    cvt_bf16_kernel<<<nw / 512, 256, 0, stream>>>(wv, Wv, nw / 2);
    cvt_bf16_kernel<<<nw / 512, 256, 0, stream>>>(wo, Wo, nw / 2);
  }

  // Stage 1: projections (scale 1/sqrt(Dh)=0.125 folded into Q)
  {
    dim3 g(BM / 128, DMODEL / 128);
    proj_qkv_kernel<<<g, 256, 0, stream>>>(Xq, Wq, Qh, 0.125f);
    proj_qkv_kernel<<<g, 256, 0, stream>>>(Xk, Wk, Kh, 1.0f);
    proj_qkv_kernel<<<g, 256, 0, stream>>>(Xv, Wv, Vh, 1.0f);
  }

  // Stage 2: flash attention (TDM-staged K panels)
  {
    dim3 g(MDIM / 128, BDIM * NHEAD);
    flash_attn_kernel<<<g, 128, 0, stream>>>(Qh, Kh, Vh, mask, Cx);
  }

  // Stage 3: output projection + residual
  {
    dim3 g(BM / 128, DMODEL / 128);
    oproj_kernel<<<g, 256, 0, stream>>>(Cx, Wo, inQ, out);
  }

  // Stage 4: LayerNorm in place
  layernorm_kernel<<<BM, 256, 0, stream>>>(out, gamma, beta);
}
